// SparseAttention_27874337751552
// MI455X (gfx1250) — matrix operand order repaired and checked by reading, not compiled, not run on hardware
//
#include <hip/hip_runtime.h>
#include <hip/hip_bf16.h>
#include <math.h>

// Sliding-window attention, |i-j| <= WIN-1, for MI455X (gfx1250, wave32).
// Flash-attention style: per-wave 16-row query tile, 32-key tiles staged in
// LDS by the whole 4-wave block (K row-major bf16, V transposed bf16),
// v_wmma_f32_16x16x32_bf16 for both QK^T and P*V, f32 online softmax with
// DPP16 row reductions (no LDS round-trips for shuffles).

typedef __attribute__((ext_vector_type(16))) __bf16 v16bf;
typedef __attribute__((ext_vector_type(8)))  float  v8f;

#define S_LEN   4096
#define D_DIM   128
#define WIN     128
#define MTILE   16
#define NTILE   32
#define WAVES   4
#define BLOCK_Q (MTILE * WAVES)   // 64 query rows per block
#define KPAD    136               // K tile row stride (ushorts), bank-skewed
#define VPAD    40                // V^T row stride (ushorts), bank-skewed
#define PPAD    40                // P buffer row stride (ushorts)

// Packed f32x2 -> bf16x2 (RNE); lowers to V_CVT_PK_BF16_F32 when available.
__device__ __forceinline__ unsigned pk2bf(float x, float y) {
  union { __hip_bfloat162 h; unsigned u; } cv;
  cv.h = __float22bfloat162_rn(make_float2(x, y));
  return cv.u;
}
__device__ __forceinline__ unsigned short f2bf(float f) {
  union { __hip_bfloat16 h; unsigned short u; } cv;
  cv.h = __float2bfloat16(f);
  return cv.u;
}

union Frag16 {
  v16bf v;
  uint4 q[2];
  unsigned int w[8];
  unsigned short h[16];
};

// ---- DPP16 reductions across the 16 lanes of a half-wave ----------------
// ctrl 0xB1 = quad_perm [1,0,3,2] (xor1), 0x4E = quad_perm [2,3,0,1] (xor2),
// 0x141 = ROW_HALF_MIRROR (xor7 on low 3 bits), 0x140 = ROW_MIRROR (xor15).
// {1,2,7,15} generates the full 16-lane group.
template <int CTRL>
__device__ __forceinline__ float dpp_step_max(float x) {
  int xi = __float_as_int(x);
  int t = __builtin_amdgcn_update_dpp(xi, xi, CTRL, 0xF, 0xF, false);
  return fmaxf(x, __int_as_float(t));
}
__device__ __forceinline__ float row16_max(float x) {
  x = dpp_step_max<0xB1>(x);
  x = dpp_step_max<0x4E>(x);
  x = dpp_step_max<0x141>(x);
  x = dpp_step_max<0x140>(x);
  return x;
}
template <int CTRL>
__device__ __forceinline__ float dpp_step_add(float x) {
  int xi = __float_as_int(x);
  int t = __builtin_amdgcn_update_dpp(xi, xi, CTRL, 0xF, 0xF, false);
  return x + __int_as_float(t);
}
__device__ __forceinline__ float row16_sum(float x) {
  x = dpp_step_add<0xB1>(x);
  x = dpp_step_add<0x4E>(x);
  x = dpp_step_add<0x141>(x);
  x = dpp_step_add<0x140>(x);
  return x;
}

// Stage one 32x128 K tile (bf16, row-major) and V tile (bf16, transposed)
// into LDS cooperatively by all 128 threads.
__device__ __forceinline__ void stage_tile(const float* __restrict__ Kb,
                                           const float* __restrict__ Vb,
                                           int j0,
                                           unsigned short (*Ks)[KPAD],
                                           unsigned short (*Vt)[VPAD],
                                           int tid) {
#pragma unroll
  for (int it = 0; it < 8; ++it) {
    int idx = it * 128 + tid;        // 0..1023 -> 32 rows x 32 float4
    int row = idx >> 5;
    int col = (idx & 31) << 2;
    const float4 kk = *(const float4*)(Kb + (size_t)(j0 + row) * D_DIM + col);
    *(uint2*)&Ks[row][col] = make_uint2(pk2bf(kk.x, kk.y), pk2bf(kk.z, kk.w));
    const float4 vv = *(const float4*)(Vb + (size_t)(j0 + row) * D_DIM + col);
    unsigned v01 = pk2bf(vv.x, vv.y);
    unsigned v23 = pk2bf(vv.z, vv.w);
    Vt[col + 0][row] = (unsigned short)(v01 & 0xFFFFu);
    Vt[col + 1][row] = (unsigned short)(v01 >> 16);
    Vt[col + 2][row] = (unsigned short)(v23 & 0xFFFFu);
    Vt[col + 3][row] = (unsigned short)(v23 >> 16);
  }
}

__global__ __launch_bounds__(128)
void swa_wmma_kernel(const float* __restrict__ Q, const float* __restrict__ K,
                     const float* __restrict__ V, float* __restrict__ O) {
  __shared__ __align__(16) unsigned short Ks[2][NTILE][KPAD];
  __shared__ __align__(16) unsigned short Vt[2][D_DIM][VPAD];
  __shared__ __align__(16) unsigned short Pb[WAVES][MTILE][PPAD];

  const int tid  = threadIdx.x;
  const int wave = tid >> 5;
  const int lane = tid & 31;
  const int hf   = lane >> 4;      // lane-half (wave32 WMMA layout)
  const int lm   = lane & 15;

  const int qb = blockIdx.x * BLOCK_Q;
  const int b  = blockIdx.y;
  const int i0 = qb + wave * MTILE;

  const float* Qb = Q + (size_t)b * S_LEN * D_DIM;
  const float* Kb = K + (size_t)b * S_LEN * D_DIM;
  const float* Vb = V + (size_t)b * S_LEN * D_DIM;
  float*       Ob = O + (size_t)b * S_LEN * D_DIM;

  // ---- Q fragments: 4 chunks of 16x32 bf16 A-matrix, scaled by 1/sqrt(D) ----
  const float scale = 0.08838834764831845f;
  Frag16 qf[4];
  {
    const float* qrow = Qb + (size_t)(i0 + lm) * D_DIM;  // A: M = lane%16
#pragma unroll
    for (int c = 0; c < 4; ++c) {
#pragma unroll
      for (int vv = 0; vv < 8; ++vv) {
        // 16-bit A layout: VGPR v<4 -> K = 2v + 8*half(+bit); v>=4 -> 16+2(v-4)+8*half
        int d = 32 * c + ((vv < 4) ? (2 * vv + 8 * hf)
                                   : (16 + 2 * (vv - 4) + 8 * hf));
        float2 f2 = *(const float2*)(qrow + d);
        qf[c].w[vv] = pk2bf(f2.x * scale, f2.y * scale);
      }
    }
  }

  // ---- flash-attention state ----
  v8f oacc[8];
#pragma unroll
  for (int dc = 0; dc < 8; ++dc) oacc[dc] = {};
  float mrow[8], lrow[8];
#pragma unroll
  for (int r = 0; r < 8; ++r) { mrow[r] = -INFINITY; lrow[r] = 0.f; }

  const int jwlo = max(0, i0 - (WIN - 1));
  const int jwhi = min(S_LEN, i0 + MTILE + (WIN - 1));
  const int jblo = max(0, qb - (WIN - 1));
  const int jbhi = min(S_LEN, qb + BLOCK_Q + (WIN - 1));
  const int t0 = jblo / NTILE;
  const int t1 = (jbhi + NTILE - 1) / NTILE;

  stage_tile(Kb, Vb, t0 * NTILE, Ks[t0 & 1], Vt[t0 & 1], tid);

  for (int t = t0; t < t1; ++t) {
    __syncthreads();  // stage of buf[t&1] complete; prev compute drained
    if (t + 1 < t1)
      stage_tile(Kb, Vb, (t + 1) * NTILE, Ks[(t + 1) & 1], Vt[(t + 1) & 1], tid);
    if (t + 2 < t1) {  // speculative prefetch of tile-after-next (global_prefetch)
      __builtin_prefetch(Kb + (size_t)(t + 2) * NTILE * D_DIM + tid * 32, 0, 0);
      __builtin_prefetch(Vb + (size_t)(t + 2) * NTILE * D_DIM + tid * 32, 0, 0);
    }

    const int j0 = t * NTILE;
    // Wave-uniform band test; force a *scalar* branch so EXEC stays all-1s
    // around WMMA (ISA requirement) instead of exec-mask predication.
    const int activeV = (j0 < jwhi) && (j0 + NTILE > jwlo);
    if (__builtin_amdgcn_readfirstlane(activeV)) {
      const int bf = t & 1;

      // ---- S = Q K^T : preload all 8 B-fragments, then 8 WMMAs ----
      Frag16 kf[8];  // [n*4+c]; B layout: N=lane%16 (key j0+16n+N), K = 8*half+e (e<8), 16+8*half+(e-8) (e>=8)
#pragma unroll
      for (int n = 0; n < 2; ++n)
#pragma unroll
        for (int c = 0; c < 4; ++c) {
          const uint4* src = (const uint4*)&Ks[bf][16 * n + lm][32 * c + 8 * hf];
          kf[n * 4 + c].q[0] = src[0];
          kf[n * 4 + c].q[1] = src[2];
        }
      v8f sc[2];
#pragma unroll
      for (int n = 0; n < 2; ++n) {
        v8f acc = {};
#pragma unroll
        for (int c = 0; c < 4; ++c)
          acc = __builtin_amdgcn_wmma_f32_16x16x32_bf16(
              false, qf[c].v, false, kf[n * 4 + c].v, (short)0, acc, false, false);
        sc[n] = acc;
      }

      // ---- band mask + online softmax (C layout: M=r+8*half, N=lane%16) ----
      float al[8];
#pragma unroll
      for (int r = 0; r < 8; ++r) {
        const int i = i0 + r + 8 * hf;
#pragma unroll
        for (int n = 0; n < 2; ++n) {
          int dji = i - (j0 + 16 * n + lm);
          if ((unsigned)(dji + (WIN - 1)) > (unsigned)(2 * WIN - 2))
            sc[n][r] = -INFINITY;
        }
        float tmax = row16_max(fmaxf(sc[0][r], sc[1][r]));
        float mnew = fmaxf(mrow[r], tmax);
        float mg = (mnew == -INFINITY) ? 0.f : mnew;
        float a = (mrow[r] == -INFINITY)
                      ? 0.f
                      : exp2f((mrow[r] - mnew) * 1.44269504f);
        mrow[r] = mnew;
        al[r] = a;
        float p0 = exp2f((sc[0][r] - mg) * 1.44269504f);
        float p1 = exp2f((sc[1][r] - mg) * 1.44269504f);
        float rs = row16_sum(p0 + p1);
        lrow[r] = lrow[r] * a + rs;
        Pb[wave][r + 8 * hf][lm]      = f2bf(p0);
        Pb[wave][r + 8 * hf][16 + lm] = f2bf(p1);
      }

      // rescale running output by alpha
#pragma unroll
      for (int dc = 0; dc < 8; ++dc)
#pragma unroll
        for (int r = 0; r < 8; ++r) oacc[dc][r] *= al[r];

      __builtin_amdgcn_wave_barrier();  // intra-wave LDS ops are in-order; fence compiler

      // ---- P fragment (C layout -> A layout via LDS round trip) ----
      Frag16 pf;
      pf.q[0] = *(const uint4*)&Pb[wave][lm][16 * hf];
      pf.q[1] = *(const uint4*)&Pb[wave][lm][16 * hf + 8];

      // ---- O += P * V : preload all 8 B-fragments, then 8 WMMAs ----
      Frag16 vf[8];
#pragma unroll
      for (int dc = 0; dc < 8; ++dc) {
        const uint4* vsrc = (const uint4*)&Vt[bf][16 * dc + lm][16 * hf];
        vf[dc].q[0] = vsrc[0];
        vf[dc].q[1] = vsrc[1];
      }
#pragma unroll
      for (int dc = 0; dc < 8; ++dc)
        oacc[dc] = __builtin_amdgcn_wmma_f32_16x16x32_bf16(
            false, pf.v, false, vf[dc].v, (short)0, oacc[dc], false, false);
    }
    __syncthreads();  // all waves done reading buf[t&1] before it is restaged
  }

  // ---- epilogue: O / l ----
#pragma unroll
  for (int r = 0; r < 8; ++r) {
    float inv = (lrow[r] > 0.f) ? (1.f / lrow[r]) : 0.f;
    float* orow = Ob + (size_t)(i0 + r + 8 * hf) * D_DIM;
#pragma unroll
    for (int dc = 0; dc < 8; ++dc) orow[16 * dc + lm] = oacc[dc][r] * inv;
  }
}

extern "C" void kernel_launch(void* const* d_in, const int* in_sizes, int n_in,
                              void* d_out, int out_size, void* d_ws, size_t ws_size,
                              hipStream_t stream) {
  const float* q = (const float*)d_in[0];
  const float* k = (const float*)d_in[1];
  const float* v = (const float*)d_in[2];
  float* out = (float*)d_out;
  const int B = in_sizes[0] / (S_LEN * D_DIM);
  dim3 grid(S_LEN / BLOCK_Q, B);
  dim3 block(128);
  hipLaunchKernelGGL(swa_wmma_kernel, grid, block, 0, stream, q, k, v, out);
}
